// MultiHeadLatentAttention_62414464745722
// MI455X (gfx1250) — matrix operand order repaired and checked by reading, not compiled, not run on hardware
//
#include <hip/hip_runtime.h>
#include <hip/hip_bf16.h>

// ---------------------------------------------------------------------------
// MLA forward for MI455X (gfx1250, wave32):
//   - WMMA bf16 16x16x32, f32 accumulate
//   - GLOBAL_LOAD_ASYNC_TO_LDS_B128 double-buffered tile staging (ASYNCcnt)
//   - fused flash attention (no [B,H,N,N] materialization)
// ---------------------------------------------------------------------------

typedef __attribute__((ext_vector_type(16))) __bf16          v16bf;
typedef __attribute__((ext_vector_type(8)))  float           v8f;
typedef __attribute__((ext_vector_type(16))) unsigned short  v16us;
typedef __attribute__((ext_vector_type(8)))  unsigned short  v8us;

#define MLA_B    4
#define MLA_N    2048
#define MLA_D    1024
#define MLA_H    16
#define MLA_HD   64
#define MLA_RANK 256
#define MLA_BN   (MLA_B * MLA_N)   // 8192

// s_wait_asynccnt <= n  (ASYNCcnt tracks async-to-LDS loads, per ISA ch.15.5)
#define WAIT_ASYNC(n) asm volatile("s_wait_asynccnt " #n ::: "memory")

// One 16-byte async global->LDS copy per lane (per-wave: 1 instruction).
__device__ __forceinline__ void async_b128(unsigned lds_byte_addr,
                                           unsigned global_voffset,
                                           unsigned long long saddr) {
  asm volatile("global_load_async_to_lds_b128 %0, %1, %2"
               :: "v"(lds_byte_addr), "v"(global_voffset), "s"(saddr)
               : "memory");
}

// ---- f32 -> bf16 (round to nearest even) ----------------------------------
__device__ __forceinline__ unsigned short f2bf(float f) {
  union { float f; unsigned int u; } v; v.f = f;
  unsigned int u = v.u;
  u += 0x7fffu + ((u >> 16) & 1u);
  return (unsigned short)(u >> 16);
}

// ---- WMMA fragment loaders (contiguous-k order, same in A and B) ---------------------------
// A 16x32 bf16: lane L holds row M=L&15; K = hi*16..+15 contiguous,
// hi = L>>4.  Works for global or LDS pointers.
__device__ __forceinline__ v16bf load_afrag(const unsigned short* row, int hi16) {
  v8us lo = *(const v8us*)(row + hi16 * 16);
  v8us hi = *(const v8us*)(row + 8 + hi16 * 16);
  v16us u;
#pragma unroll
  for (int i = 0; i < 8; ++i) { u[i] = lo[i]; u[i + 8] = hi[i]; }
  union { v16us u; v16bf b; } cv; cv.u = u;
  return cv.b;
}
// B 32x16 bf16: lane L holds column N=L&15, K = (L>>4)*16..+15 contiguous.
__device__ __forceinline__ v16bf load_bfrag(const unsigned short* p) {
  v8us lo = *(const v8us*)(p);
  v8us hi = *(const v8us*)(p + 8);
  v16us u;
#pragma unroll
  for (int i = 0; i < 8; ++i) { u[i] = lo[i]; u[i + 8] = hi[i]; }
  union { v16us u; v16bf b; } cv; cv.u = u;
  return cv.b;
}

__device__ __forceinline__ v8f wmma_bf16(v16bf a, v16bf b, v8f c) {
  return __builtin_amdgcn_wmma_f32_16x16x32_bf16(
      false, a, false, b, (short)0, c, false, false);
}

// ---------------------------------------------------------------------------
// Kernel 0: f32 -> bf16 cast (grid-stride)
// ---------------------------------------------------------------------------
__global__ void cast_bf16_kernel(const float* __restrict__ in,
                                 unsigned short* __restrict__ out,
                                 long long n) {
  long long i = (long long)blockIdx.x * blockDim.x + threadIdx.x;
  long long stride = (long long)gridDim.x * blockDim.x;
  for (; i < n; i += stride) {
    __builtin_prefetch(in + i + 8192, 0, 1);
    out[i] = f2bf(in[i]);
  }
}

// ---------------------------------------------------------------------------
// Kernel 1: C[M,Nout] = A[M,K](bf16) @ Bw[Nout,K](bf16)^T + bias, * scale
//   mode 0: bf16 [M,Nout] | mode 1: bf16 V-transposed [B,H,HD,N] | mode 2: f32
// Block: 256 thr = 8 waves (4 row-groups x 2 col-groups); tile 128x128,
// K-step 32, double-buffered async LDS staging.
// LDS rows padded to 40 elems (80 B) to spread banks for ds_load_b128.
// ---------------------------------------------------------------------------
__global__ __launch_bounds__(256) void gemm_bf16_kernel(
    const unsigned short* __restrict__ A,
    const unsigned short* __restrict__ Bw,
    const float* __restrict__ bias,
    void* __restrict__ Cout,
    int M, int Nout, int K, float scale, int mode) {
  __shared__ unsigned short smem[4 * 128 * 40];  // [A0|A1|B0|B1] = 40 KB

  const int t    = threadIdx.x;
  const int lane = t & 31;
  const int wave = t >> 5;
  const int g    = lane & 15;
  const int hi16 = lane >> 4;
  const int wr   = wave >> 1;       // 0..3  -> rows wr*32..+31
  const int wc   = wave & 1;        // 0..1  -> cols wc*64..+63
  const int m0 = blockIdx.x * 128;
  const int n0 = blockIdx.y * 128;
  const unsigned ldsBase = (unsigned)(size_t)(void*)smem;

  v8f acc[2][4] = {};

  // stage one K-step: A tile 128x32 (64 B rows) + B tile 128x32.
  // 512 16-B chunks per tile; 256 threads x 2 chunks; 4 async instr / wave.
  auto issue = [&](int buf, int k0) {
    const unsigned long long sa =
        (unsigned long long)(A + ((long long)m0 * K + k0));
    const unsigned long long sb =
        (unsigned long long)(Bw + ((long long)n0 * K + k0));
    const unsigned la = ldsBase + (unsigned)buf * 10240u;
    const unsigned lb = ldsBase + 20480u + (unsigned)buf * 10240u;
    const int rs = K * 2;  // global row stride in bytes
#pragma unroll
    for (int i = 0; i < 2; ++i) {
      const int c = t + i * 256;
      const int row = c >> 2, kp = c & 3;
      const unsigned voff = (unsigned)(row * rs + kp * 16);
      const unsigned loff = (unsigned)(row * 80 + kp * 16);
      async_b128(la + loff, voff, sa);
      async_b128(lb + loff, voff, sb);
    }
  };

  issue(0, 0);
  int buf = 0;
  for (int k0 = 0; k0 < K; k0 += 32) {
    const bool more = (k0 + 32) < K;
    if (more) {
      issue(buf ^ 1, k0 + 32);
      WAIT_ASYNC(4);       // current buffer's 4 copies done; next 4 in flight
    } else {
      WAIT_ASYNC(0);
    }
    __syncthreads();       // whole block's copies into `buf` complete

    const unsigned short* Ab = smem + buf * 5120;
    const unsigned short* Bb = smem + 10240 + buf * 5120;
    v16bf afr[2], bfr[4];
#pragma unroll
    for (int ms = 0; ms < 2; ++ms)
      afr[ms] = load_afrag(Ab + (wr * 32 + ms * 16 + g) * 40, hi16);
#pragma unroll
    for (int s = 0; s < 4; ++s)
      bfr[s] = load_bfrag(Bb + (wc * 64 + s * 16 + g) * 40 + hi16 * 16);
#pragma unroll
    for (int ms = 0; ms < 2; ++ms)
#pragma unroll
      for (int s = 0; s < 4; ++s)
        acc[ms][s] = wmma_bf16(afr[ms], bfr[s], acc[ms][s]);

    __syncthreads();       // everyone done reading `buf` before it is refilled
    buf ^= 1;
  }

#pragma unroll
  for (int ms = 0; ms < 2; ++ms) {
#pragma unroll
    for (int s = 0; s < 4; ++s) {
      const int n = n0 + wc * 64 + s * 16 + g;
      const float bv = bias ? bias[n] : 0.0f;
#pragma unroll
      for (int r = 0; r < 8; ++r) {
        const int m = m0 + wr * 32 + ms * 16 + hi16 * 8 + r;
        const float v = (acc[ms][s][r] + bv) * scale;
        if (mode == 2) {
          ((float*)Cout)[(long long)m * Nout + n] = v;
        } else if (mode == 0) {
          ((unsigned short*)Cout)[(long long)m * Nout + n] = f2bf(v);
        } else {  // V-transposed: [B, H, HD, N]
          const int b = m >> 11, tok = m & 2047;
          const int h = n >> 6,  d   = n & 63;
          ((unsigned short*)Cout)
              [(((long long)(b * MLA_H + h) * MLA_HD + d) << 11) + tok] =
              f2bf(v);
        }
      }
    }
  }
}

// ---------------------------------------------------------------------------
// Kernel 2: fused flash attention per (b,h).
//   Qb [B*N,D] bf16 (pre-scaled by 1/8), Kb [B*N,D] bf16,
//   Vt [B,H,HD,N] bf16, Ob [B*N,D] bf16.
// Block 256 thr = 8 waves; each wave owns 16 q-rows; key tiles of 32.
// K tile (32x64, rows padded to 72) and V tile (64x32, rows padded to 40)
// double-buffered in LDS via async copies, shared by all 8 waves.
// ---------------------------------------------------------------------------
__global__ __launch_bounds__(256) void mla_attn_kernel(
    const unsigned short* __restrict__ Qb,
    const unsigned short* __restrict__ Kb,
    const unsigned short* __restrict__ Vt,
    unsigned short* __restrict__ Ob) {
  __shared__ unsigned short kvs[2 * 32 * 72 + 2 * 64 * 40];  // K bufs | V bufs
  __shared__ unsigned short pbuf[8 * 16 * 32];               // per-wave P tile

  const int t    = threadIdx.x;
  const int lane = t & 31;
  const int wave = t >> 5;
  const int g    = lane & 15;
  const int hi16 = lane >> 4;
  const int bh = blockIdx.x;            // 0..63
  const int b  = bh >> 4, h = bh & 15;
  const int q0 = blockIdx.y * 128 + wave * 16;
  const unsigned kvBase = (unsigned)(size_t)(void*)kvs;

  // K tile: 32 rows x 128 B (row = key token). V tile: 64 rows x 64 B (row=d).
  auto issueKV = [&](int buf, int j0) {
    const unsigned long long sk = (unsigned long long)(
        Kb + ((long long)(b * MLA_N + j0) * MLA_D + h * MLA_HD));
    const unsigned long long sv = (unsigned long long)(
        Vt + (((long long)bh * MLA_HD) << 11) + j0);
    const unsigned lk = kvBase + (unsigned)buf * 4608u;            // 32*72*2
    const unsigned lv = kvBase + 9216u + (unsigned)buf * 5120u;    // 64*40*2
    {
      const int row = t >> 3, kp = t & 7;        // 256 chunks of 16 B
      async_b128(lk + (unsigned)(row * 144 + kp * 16),
                 (unsigned)(row * (MLA_D * 2) + kp * 16), sk);
    }
    {
      const int row = t >> 2, kp = t & 3;        // 256 chunks of 16 B
      async_b128(lv + (unsigned)(row * 80 + kp * 16),
                 (unsigned)(row * (MLA_N * 2) + kp * 16), sv);
    }
  };

  // Q A-fragments (HD=64 -> two K=32 steps), loaded once from global.
  v16bf aq[2];
  {
    const unsigned short* qrow =
        Qb + (long long)(b * MLA_N + q0 + g) * MLA_D + h * MLA_HD;
    aq[0] = load_afrag(qrow, hi16);
    aq[1] = load_afrag(qrow + 32, hi16);
  }

  float mrun[8], lrun[8];
  v8f o[4] = {};
#pragma unroll
  for (int r = 0; r < 8; ++r) { mrun[r] = -1e30f; lrun[r] = 0.0f; }

  unsigned short* myP = pbuf + wave * (16 * 32);

  issueKV(0, 0);
  int buf = 0;
  for (int j0 = 0; j0 < MLA_N; j0 += 32) {
    const bool more = (j0 + 32) < MLA_N;
    if (more) {
      issueKV(buf ^ 1, j0 + 32);
      WAIT_ASYNC(2);
    } else {
      WAIT_ASYNC(0);
    }
    __syncthreads();   // K/V tiles for `buf` visible to all waves

    const unsigned short* Kc = kvs + buf * 2304;
    const unsigned short* Vc = kvs + 4608 + buf * 2560;

    // ---- S = Q K^T for keys [j0, j0+32) --------------------------------
    v8f s0 = {}, s1 = {};
#pragma unroll
    for (int kk = 0; kk < 2; ++kk) {
      s0 = wmma_bf16(aq[kk],
                     load_bfrag(Kc + g * 72 + kk * 32 + hi16 * 16), s0);
      s1 = wmma_bf16(aq[kk],
                     load_bfrag(Kc + (16 + g) * 72 + kk * 32 + hi16 * 16), s1);
    }

    // ---- online softmax -------------------------------------------------
    float p0[8], p1[8], alpha[8];
#pragma unroll
    for (int r = 0; r < 8; ++r) {
      float v = fmaxf(s0[r], s1[r]);
#pragma unroll
      for (int off = 8; off >= 1; off >>= 1)
        v = fmaxf(v, __shfl_xor(v, off, 32));   // within lane group of 16
      const float mnew = fmaxf(mrun[r], v);
      const float a = __expf(mrun[r] - mnew);
      p0[r] = __expf(s0[r] - mnew);
      p1[r] = __expf(s1[r] - mnew);
      float rs = p0[r] + p1[r];
#pragma unroll
      for (int off = 8; off >= 1; off >>= 1)
        rs += __shfl_xor(rs, off, 32);
      lrun[r] = lrun[r] * a + rs;
      mrun[r] = mnew;
      alpha[r] = a;
    }
#pragma unroll
    for (int s4 = 0; s4 < 4; ++s4)
#pragma unroll
      for (int r = 0; r < 8; ++r) o[s4][r] *= alpha[r];

    // ---- P (C/D layout) -> per-wave LDS slice -> A fragment -------------
    // Same-wave DS ops are in-order (ISA 15.5): no barrier needed here.
#pragma unroll
    for (int r = 0; r < 8; ++r) {
      myP[(hi16 * 8 + r) * 32 + g]      = f2bf(p0[r]);
      myP[(hi16 * 8 + r) * 32 + 16 + g] = f2bf(p1[r]);
    }
    v16bf ap = load_afrag(myP + g * 32, hi16);

    // ---- O += P V (B-fragments from token-contiguous V tile in LDS) -----
#pragma unroll
    for (int s4 = 0; s4 < 4; ++s4)
      o[s4] = wmma_bf16(ap,
                        load_bfrag(Vc + (s4 * 16 + g) * 40 + hi16 * 16),
                        o[s4]);

    __syncthreads();   // all waves done reading `buf` before refill
    buf ^= 1;
  }

  // ---- normalize and store (heads re-concatenated into [BN, D]) ---------
#pragma unroll
  for (int s4 = 0; s4 < 4; ++s4) {
#pragma unroll
    for (int r = 0; r < 8; ++r) {
      const int qtok = q0 + hi16 * 8 + r;
      const float v = o[s4][r] / lrun[r];
      Ob[(long long)(b * MLA_N + qtok) * MLA_D + h * MLA_HD + s4 * 16 + g] =
          f2bf(v);
    }
  }
}

// ---------------------------------------------------------------------------
// Host-side orchestration
// ---------------------------------------------------------------------------
extern "C" void kernel_launch(void* const* d_in, const int* in_sizes, int n_in,
                              void* d_out, int out_size, void* d_ws, size_t ws_size,
                              hipStream_t stream) {
  (void)in_sizes; (void)n_in; (void)out_size; (void)ws_size;

  const float* x  = (const float*)d_in[0];
  const float* Wq = (const float*)d_in[1];
  const float* bq = (const float*)d_in[2];
  const float* Wd = (const float*)d_in[3];
  const float* bd = (const float*)d_in[4];
  const float* Wk = (const float*)d_in[5];
  const float* bk = (const float*)d_in[6];
  const float* Wv = (const float*)d_in[7];
  const float* bv = (const float*)d_in[8];
  const float* Wo = (const float*)d_in[9];
  const float* bo = (const float*)d_in[10];
  float* out = (float*)d_out;

  char* ws = (char*)d_ws;
  size_t off = 0;
  auto alloc = [&](size_t bytes) -> unsigned short* {
    unsigned short* p = (unsigned short*)(ws + off);
    off += (bytes + 255) & ~(size_t)255;
    return p;
  };

  const size_t BN = MLA_BN;
  unsigned short* Xb  = alloc(BN * MLA_D * 2);
  unsigned short* Qb  = alloc(BN * MLA_D * 2);
  unsigned short* KVb = alloc(BN * MLA_RANK * 2);
  unsigned short* Kb  = alloc(BN * MLA_D * 2);
  unsigned short* Vt  = alloc(BN * MLA_D * 2);
  unsigned short* Ob  = alloc(BN * MLA_D * 2);
  unsigned short* Wqb = alloc((size_t)MLA_D * MLA_D * 2);
  unsigned short* Wdb = alloc((size_t)MLA_RANK * MLA_D * 2);
  unsigned short* Wkb = alloc((size_t)MLA_D * MLA_RANK * 2);
  unsigned short* Wvb = alloc((size_t)MLA_D * MLA_RANK * 2);
  unsigned short* Wob = alloc((size_t)MLA_D * MLA_D * 2);

  cast_bf16_kernel<<<1024, 256, 0, stream>>>(x,  Xb,  (long long)BN * MLA_D);
  cast_bf16_kernel<<<256,  256, 0, stream>>>(Wq, Wqb, (long long)MLA_D * MLA_D);
  cast_bf16_kernel<<<128,  256, 0, stream>>>(Wd, Wdb, (long long)MLA_RANK * MLA_D);
  cast_bf16_kernel<<<128,  256, 0, stream>>>(Wk, Wkb, (long long)MLA_D * MLA_RANK);
  cast_bf16_kernel<<<128,  256, 0, stream>>>(Wv, Wvb, (long long)MLA_D * MLA_RANK);
  cast_bf16_kernel<<<256,  256, 0, stream>>>(Wo, Wob, (long long)MLA_D * MLA_D);

  const dim3 blk(256);
  const float qscale = 0.125f;  // 1/sqrt(HD=64), folded into Q

  // Q = x @ Wq^T (scaled)
  gemm_bf16_kernel<<<dim3(64, 8), blk, 0, stream>>>(
      Xb, Wqb, bq, Qb, MLA_BN, MLA_D, MLA_D, qscale, 0);
  // KV latent = x @ Wd^T
  gemm_bf16_kernel<<<dim3(64, 2), blk, 0, stream>>>(
      Xb, Wdb, bd, KVb, MLA_BN, MLA_RANK, MLA_D, 1.0f, 0);
  // K = KV @ Wk^T
  gemm_bf16_kernel<<<dim3(64, 8), blk, 0, stream>>>(
      KVb, Wkb, bk, Kb, MLA_BN, MLA_D, MLA_RANK, 1.0f, 0);
  // V = KV @ Wv^T, stored transposed [B,H,HD,N]
  gemm_bf16_kernel<<<dim3(64, 8), blk, 0, stream>>>(
      KVb, Wvb, bv, Vt, MLA_BN, MLA_D, MLA_RANK, 1.0f, 1);

  // fused flash attention
  mla_attn_kernel<<<dim3(MLA_B * MLA_H, MLA_N / 128), blk, 0, stream>>>(
      Qb, Kb, Vt, Ob);

  // out = O @ Wo^T + bo (f32)
  gemm_bf16_kernel<<<dim3(64, 8), blk, 0, stream>>>(
      Ob, Wob, bo, out, MLA_BN, MLA_D, MLA_D, 1.0f, 2);
}
